// RPN_60507499266506
// MI455X (gfx1250) — matrix operand order repaired and checked by reading, not compiled, not run on hardware
//
#include <hip/hip_runtime.h>
#include <hip/hip_bf16.h>
#include <stdint.h>

// ---------------- types ----------------
typedef _Float16 h16;
typedef _Float16 v8h  __attribute__((ext_vector_type(8)));
typedef _Float16 v16h __attribute__((ext_vector_type(16)));
typedef float    v8f  __attribute__((ext_vector_type(8)));
typedef unsigned int v4u __attribute__((ext_vector_type(4)));
typedef unsigned int v8u __attribute__((ext_vector_type(8)));

#define CIN   2048
#define COUT  2048
#define HW    784            // 28*28
#define KDIM  (CIN * 9)      // 18432
#define BM    128
#define BN    112
#define BK    32
#define NSTEP (KDIM / BK)    // 576
#define NANCH 7056           // 784 * 9

// ---------------------------------------------------------------------------
// TDM issue helper: 2D tile descriptor (D# groups 0/1), rows of 64B (32 f16),
// row stride = KDIM elements. Wave-scalar operands -> SGPR groups.
// ---------------------------------------------------------------------------
__device__ __forceinline__ void tdm_load_2d(unsigned lds_addr,
                                            unsigned long long gaddr,
                                            v8u g1) {
  v4u g0;
  g0[0] = 1u;                                   // count=1 (valid), user mode
  g0[1] = lds_addr;                             // lds_addr (bytes)
  g0[2] = (unsigned)gaddr;                      // global_addr[31:0]
  g0[3] = (unsigned)(gaddr >> 32) | (2u << 30); // addr[56:32] | type=2
  asm volatile("tensor_load_to_lds %0, %1" :: "s"(g0), "s"(g1) : "memory");
}

__device__ __forceinline__ v8u tdm_group1(unsigned tile_rows, unsigned tensor_rows) {
  v8u g1;
  g1[0] = (1u << 16);                               // wg_mask=0, data_size=2B
  g1[1] = ((unsigned)(KDIM & 0xFFFF) << 16);        // tensor_dim0[15:0] @63:48
  g1[2] = ((unsigned)KDIM >> 16)                    // tensor_dim0[31:16]
        | ((tensor_rows & 0xFFFFu) << 16);          // tensor_dim1[15:0] @111:80
  g1[3] = (tensor_rows >> 16) | (32u << 16);        // dim1 hi | tile_dim0=32
  g1[4] = tile_rows;                                // tile_dim1, tile_dim2=0
  g1[5] = (unsigned)KDIM;                           // tensor_dim0_stride[31:0]
  g1[6] = 0u;
  g1[7] = 0u;
  return g1;
}

// ---------------------------------------------------------------------------
// 1a) im2col: x[2048][28][28] fp32 -> B[784 pix][18432 k] f16 (N-major so each
//     32-wide K slice of a pixel row is a contiguous 64B line = one TDM row)
// ---------------------------------------------------------------------------
__global__ void im2col_f16(const float* __restrict__ x, h16* __restrict__ B) {
  long idx = (long)blockIdx.x * blockDim.x + threadIdx.x;
  const long total = (long)HW * KDIM;
  if (idx >= total) return;
  int pix = (int)(idx / KDIM);
  int k   = (int)(idx % KDIM);
  int c = k / 9, rem = k % 9;
  int kh = rem / 3, kw = rem % 3;
  int y = pix / 28, xp = pix % 28;
  int yy = y + kh - 1, xx = xp + kw - 1;
  float v = 0.0f;
  if ((unsigned)yy < 28u && (unsigned)xx < 28u)
    v = x[c * HW + yy * 28 + xx];
  B[(long)pix * KDIM + k] = (h16)v;
}

// ---------------------------------------------------------------------------
// 1b) one-time weight convert: W fp32 [2048][18432] -> f16 (75.5 MB, L2-resident)
// ---------------------------------------------------------------------------
__global__ void cvt_w_f16(const float* __restrict__ W, h16* __restrict__ Wh) {
  long idx = ((long)blockIdx.x * blockDim.x + threadIdx.x) * 8;
  if (idx >= (long)COUT * KDIM) return;
  float4 f0 = *(const float4*)(W + idx);
  float4 f1 = *(const float4*)(W + idx + 4);
  v8h o;
  o[0] = (h16)f0.x; o[1] = (h16)f0.y; o[2] = (h16)f0.z; o[3] = (h16)f0.w;
  o[4] = (h16)f1.x; o[5] = (h16)f1.y; o[6] = (h16)f1.z; o[7] = (h16)f1.w;
  *(v8h*)(Wh + idx) = o;
}

// ---------------------------------------------------------------------------
// 2) Implicit-GEMM conv, all-TDM staged, double-buffered LDS.
//    C[2048][784] = Wh[2048][18432] * B[18432][784], f16 in / f32 WMMA acc.
//    256 threads = 8 wave32; tile 128M x 112N x 32K; grid (7,16).
//    Wave w owns M-strip [w*16, w*16+16) x all 7 N-tiles.
//    Per step: wave0 DMAs next A+B tiles via tensor_load_to_lds while all
//    waves run 7 back-to-back v_wmma_f32_16x16x32_f16 on the current tiles.
// ---------------------------------------------------------------------------
__global__ __launch_bounds__(256)
void conv_gemm_wmma(const h16* __restrict__ Wh, const float* __restrict__ bias,
                    const h16* __restrict__ B, h16* __restrict__ Hout) {
  __shared__ h16 As[2][BM * BK];   // 2 x 8 KB
  __shared__ h16 Bs[2][BN * BK];   // 2 x 7 KB

  const int tid    = threadIdx.x;
  const int wave   = tid >> 5;        // 0..7 -> M-tile row
  const int lane   = tid & 31;
  const int khalf  = lane >> 4;       // 0: lanes 0-15, 1: lanes 16-31
  const int l      = lane & 15;
  const int m_base = blockIdx.y * BM;
  const int n_base = blockIdx.x * BN;

  v8f acc[7] = {};

  // uniform descriptor state
  const unsigned ldsA0 = (unsigned)(uintptr_t)&As[0][0];
  const unsigned ldsA1 = (unsigned)(uintptr_t)&As[1][0];
  const unsigned ldsB0 = (unsigned)(uintptr_t)&Bs[0][0];
  const unsigned ldsB1 = (unsigned)(uintptr_t)&Bs[1][0];
  const unsigned long long gA =
      (unsigned long long)(uintptr_t)(Wh + (long)m_base * KDIM);
  const unsigned long long gB =
      (unsigned long long)(uintptr_t)(B + (long)n_base * KDIM);
  const v8u g1A = tdm_group1(BM, COUT);  // 128-row tile of 2048-row tensor
  const v8u g1B = tdm_group1(BN, HW);    // 112-row tile of  784-row tensor

  // prologue: DMA tile 0 into buffer 0
  if (tid < 32) {
    tdm_load_2d(ldsA0, gA, g1A);
    tdm_load_2d(ldsB0, gB, g1B);
  }

  for (int s = 0; s < NSTEP; ++s) {
    const int cur = s & 1;
    __syncthreads();  // compute(s-1) done everywhere: buffer 'nxt' is free
    if (tid < 32) {
      if (s + 1 < NSTEP) {
        const unsigned long long off = (unsigned long long)(s + 1) * (BK * 2);
        tdm_load_2d(cur ? ldsA0 : ldsA1, gA + off, g1A);
        tdm_load_2d(cur ? ldsB0 : ldsB1, gB + off, g1B);
        __builtin_amdgcn_s_wait_tensorcnt(2);  // pair(s) landed, pair(s+1) flies
      } else {
        __builtin_amdgcn_s_wait_tensorcnt(0);
      }
    }
    __syncthreads();  // publish tile s to all waves

    // ---- A fragment: ISA 16-bit A layout (lanes 0-15: K0-7,16-23) ----
    const h16* ap = &As[cur][(wave * 16 + l) * BK];
    v8h a_lo = *(const v8h*)(ap + khalf * 8);        // K 0-7   / 8-15
    v8h a_hi = *(const v8h*)(ap + 16 + khalf * 8);   // K 16-23 / 24-31
    v16h afrag = __builtin_shufflevector(a_lo, a_hi,
        0, 1, 2, 3, 4, 5, 6, 7, 8, 9, 10, 11, 12, 13, 14, 15);

    // ---- preload all 7 B fragments, then 7 WMMAs back-to-back ----
    v16h bfrag[7];
    #pragma unroll
    for (int n = 0; n < 7; ++n) {
      const h16* bp = &Bs[cur][(n * 16 + l) * BK];
      v8h b0 = *(const v8h*)(bp + khalf * 8);
      v8h b1 = *(const v8h*)(bp + 16 + khalf * 8);
      bfrag[n] = __builtin_shufflevector(b0, b1,
          0, 1, 2, 3, 4, 5, 6, 7, 8, 9, 10, 11, 12, 13, 14, 15);
    }
    #pragma unroll
    for (int n = 0; n < 7; ++n)
      acc[n] = __builtin_amdgcn_wmma_f32_16x16x32_f16(
          /*neg_a=*/false, afrag, /*neg_b=*/false, bfrag[n],
          /*c_mod=*/(short)0, acc[n], /*reuse_a=*/false, /*reuse_b=*/false);
  }

  // ---- epilogue: bias + ReLU, store h as f16 [m][pix] ----
  const int mrow_off = khalf * 8;  // ISA C/D layout: lanes16-31 -> M = v+8
  #pragma unroll
  for (int n = 0; n < 7; ++n) {
    const int col = n_base + n * 16 + l;
    #pragma unroll
    for (int v = 0; v < 8; ++v) {
      const int m = m_base + wave * 16 + v + mrow_off;
      float val = acc[n][v] + bias[m];
      val = val > 0.0f ? val : 0.0f;
      Hout[(long)m * HW + col] = (h16)val;
    }
  }
}

// ---------------------------------------------------------------------------
// 3) 1x1 heads + sigmoid: RC[o][pix], o<36 = reg channels, o in [36,45) = cls
// ---------------------------------------------------------------------------
__global__ void heads_sigmoid(const h16* __restrict__ H,
                              const float* __restrict__ rw, const float* __restrict__ rb,
                              const float* __restrict__ cw, const float* __restrict__ cb,
                              float* __restrict__ RC) {
  int idx = blockIdx.x * blockDim.x + threadIdx.x;
  if (idx >= 45 * HW) return;
  int o = idx / HW, pix = idx % HW;
  const float* wrow = (o < 36) ? (rw + o * CIN) : (cw + (o - 36) * CIN);
  float s = (o < 36) ? rb[o] : cb[o - 36];
  for (int c = 0; c < CIN; ++c)
    s += wrow[c] * (float)H[(long)c * HW + pix];
  RC[idx] = 1.0f / (1.0f + __expf(-s));
}

// ---------------------------------------------------------------------------
// 4) static anchor validity mask (mirrors RPN._make_anchor_box geometry)
//    index ordering: v = pixel*9 + anchor, anchor = scale_idx*3 + ratio_idx
// ---------------------------------------------------------------------------
__global__ void anchor_mask(unsigned char* __restrict__ mask) {
  int v = blockIdx.x * blockDim.x + threadIdx.x;
  if (v >= NANCH) return;
  int pix = v / 9, a = v % 9;
  int y = pix / 28, xp = pix % 28;
  float cy = (y + 0.5f) / 28.0f;
  float cx = (xp + 0.5f) / 28.0f;
  const float scales[3] = {64.0f, 128.0f, 256.0f};
  const float ratios[3] = {0.5f, 1.0f, 2.0f};
  float s = scales[a / 3], r = ratios[a % 3];
  float sr = sqrtf(r);
  float ah = s / sr / 448.0f;
  float aw = s * sr / 448.0f;
  float y1 = cy - 0.5f * ah, x1 = cx - 0.5f * aw;
  float y2 = cy + 0.5f * ah, x2 = cx + 0.5f * aw;
  mask[v] = (y1 > 0.0f && x1 > 0.0f && y2 < 28.0f && x2 < 28.0f) ? 1 : 0;
}

// ---------------------------------------------------------------------------
// 5) rank-compact gather + threshold. Replicates the reference's
//    view/permute index math: out[j][i] = reg[9i + v/784][v%784] * keep,
//    keep = sigmoid(cls[v/784][v%784]) > 0.9, j = rank of v among valid.
// ---------------------------------------------------------------------------
__global__ void gather_emit(const unsigned char* __restrict__ mask,
                            const float* __restrict__ RC,
                            float* __restrict__ out) {
  int v = blockIdx.x * blockDim.x + threadIdx.x;
  if (v >= NANCH) return;
  if (!mask[v]) return;
  int rank = 0;
  for (int u = 0; u < v; ++u) rank += mask[u];
  int a = v / HW, pix = v % HW;
  float keep = (RC[(36 + a) * HW + pix] > 0.9f) ? 1.0f : 0.0f;
  #pragma unroll
  for (int i = 0; i < 4; ++i)
    out[rank * 4 + i] = RC[(9 * i + a) * HW + pix] * keep;
}

// ---------------------------------------------------------------------------
extern "C" void kernel_launch(void* const* d_in, const int* in_sizes, int n_in,
                              void* d_out, int out_size, void* d_ws, size_t ws_size,
                              hipStream_t stream) {
  const float* x      = (const float*)d_in[0];  // [1,2048,28,28]
  const float* conv_w = (const float*)d_in[1];  // [2048,2048,3,3] = [2048][18432]
  const float* conv_b = (const float*)d_in[2];  // [2048]
  const float* reg_w  = (const float*)d_in[3];  // [36,2048]
  const float* reg_b  = (const float*)d_in[4];  // [36]
  const float* cls_w  = (const float*)d_in[5];  // [9,2048]
  const float* cls_b  = (const float*)d_in[6];  // [9]
  float* out = (float*)d_out;                   // [4850,4] fp32

  // workspace layout
  char* ws = (char*)d_ws;
  const size_t B_BYTES  = (size_t)HW * KDIM * sizeof(h16);      // 28,901,376
  const size_t W_BYTES  = (size_t)COUT * KDIM * sizeof(h16);    // 75,497,472
  const size_t H_BYTES  = (size_t)COUT * HW * sizeof(h16);      //  3,211,264
  const size_t RC_BYTES = (size_t)45 * HW * sizeof(float);      //    141,120
  h16*  B    = (h16*)ws;
  h16*  Wh   = (h16*)(ws + B_BYTES);
  h16*  H    = (h16*)(ws + B_BYTES + W_BYTES);
  float* RC  = (float*)(ws + B_BYTES + W_BYTES + H_BYTES);
  unsigned char* mask =
      (unsigned char*)(ws + B_BYTES + W_BYTES + H_BYTES + RC_BYTES);

  const long tot = (long)HW * KDIM;
  im2col_f16<<<(int)((tot + 255) / 256), 256, 0, stream>>>(x, B);

  const long wtot = (long)COUT * KDIM;  // 37,748,736 (divisible by 8*256)
  cvt_w_f16<<<(int)(wtot / (8 * 256)), 256, 0, stream>>>(conv_w, Wh);

  dim3 grid(HW / BN, COUT / BM);  // (7, 16)
  conv_gemm_wmma<<<grid, 256, 0, stream>>>(Wh, conv_b, B, H);

  heads_sigmoid<<<(45 * HW + 255) / 256, 256, 0, stream>>>(H, reg_w, reg_b,
                                                           cls_w, cls_b, RC);
  anchor_mask<<<(NANCH + 255) / 256, 256, 0, stream>>>(mask);
  gather_emit<<<(NANCH + 255) / 256, 256, 0, stream>>>(mask, RC, out);
}
